// TC_EMNet_23656679867613
// MI455X (gfx1250) — matrix operand order repaired and checked by reading, not compiled, not run on hardware
//
#include <hip/hip_runtime.h>
#include <hip/hip_bf16.h>

#define Bb 32
#define Ss 1024
#define INd 35
#define Hd 128
#define OUTd 3

typedef __attribute__((ext_vector_type(16))) _Float16 v16h;
typedef __attribute__((ext_vector_type(8)))  _Float16 v8h;
typedef __attribute__((ext_vector_type(8)))  float    v8f;

__device__ __forceinline__ v8f wmma_f16(v16h a, v16h b, v8f c) {
    return __builtin_amdgcn_wmma_f32_16x16x32_f16(false, a, false, b, (short)0, c, false, false);
}
__device__ __forceinline__ v8f vzero8() {
    v8f v;
#pragma unroll
    for (int i = 0; i < 8; i++) v[i] = 0.f;
    return v;
}
__device__ __forceinline__ v16h cat8(v8h lo, v8h hi) {
    return __builtin_shufflevector(lo, hi, 0,1,2,3,4,5,6,7,8,9,10,11,12,13,14,15);
}
__device__ __forceinline__ float sigf(float x) { return 1.f / (1.f + __expf(-x)); }

// ---------------------------------------------------------------------------
// Tensor Data Mover staging (gfx1250): DMA a 32-row x 128-f16 2D tile from
// global to LDS, inserting 32B padding after each 256B row (-> 144-half rows).
// This toolchain exposes the 6-arg builtin: (g0, g1, g2, g3, g4, cpol).
// ---------------------------------------------------------------------------
#if __has_builtin(__builtin_amdgcn_tensor_load_to_lds)
#define USE_TDM 1
typedef __attribute__((ext_vector_type(4))) unsigned int v4u;
typedef __attribute__((ext_vector_type(8))) int v8i;
typedef __attribute__((ext_vector_type(4))) int v4i;

__device__ __forceinline__ void tdm_load_rows(const _Float16* gsrc, unsigned lds_off) {
    unsigned long long ga = (unsigned long long)(size_t)gsrc;
    v4u g0;
    g0[0] = 1u;                                              // count=1 valid descriptor
    g0[1] = lds_off;                                         // lds_addr (bytes)
    g0[2] = (unsigned)(ga & 0xFFFFFFFFu);                    // global_addr[31:0]
    g0[3] = (unsigned)((ga >> 32) & 0x1FFFFFFu) | (2u << 30);// global_addr[56:32] | type=2
    v8i g1;
    g1[0] = (int)((1u << 16) | (1u << 20) | (5u << 22) | (7u << 25));
    //            data=2B      pad_en       64DW interval   8DW pad
    g1[1] = (int)(128u << 16);         // tensor_dim0 = 128 elems  ([63:48])
    g1[2] = (int)(1024u << 16);        // tensor_dim1 = S          ([95:80])
    g1[3] = (int)(128u << 16);         // tile_dim0 = 128          ([127:112])
    g1[4] = 32;                        // tile_dim1 = 32 rows      ([143:128])
    g1[5] = 128;                       // tensor_dim0_stride = H   ([207:160])
    g1[6] = 0;
    g1[7] = 0;
    v4i gz4; gz4[0] = gz4[1] = gz4[2] = gz4[3] = 0;
    v8i gz8;
#pragma unroll
    for (int i = 0; i < 8; i++) gz8[i] = 0;
    __builtin_amdgcn_tensor_load_to_lds(g0, g1, gz4, gz4, gz8, 0);
}
#else
#define USE_TDM 0
#endif

// ---------------------------------------------------------------------------
// WMMA GEMM, all-f16 operands, padded buffers -> branch-free inner loop.
// ---------------------------------------------------------------------------
__global__ __launch_bounds__(128) void wmma_gemm_kernel(
    const _Float16* __restrict__ A, int lda,
    const _Float16* __restrict__ Wt, int ldb,
    const float* __restrict__ bias,
    float* __restrict__ C, _Float16* __restrict__ Ch,
    int Mm, int Nn, int Kpad, int act)
{
    int lane = threadIdx.x & 31;
    int wv   = threadIdx.x >> 5;
    int tiles_n = (Nn + 15) >> 4;
    int tg = blockIdx.x * 4 + wv;
    if (tg >= (Mm >> 4) * tiles_n) return;
    int m0 = (tg / tiles_n) << 4;
    int n0 = (tg % tiles_n) << 4;
    int mrow = m0 + (lane & 15);
    int ncol = n0 + (lane & 15);
    const _Float16* Ap = A  + (size_t)mrow * lda + ((lane >> 4) << 4);
    const _Float16* Bp = Wt + (size_t)ncol * ldb + ((lane >> 4) << 4);

    v8f acc = vzero8();
    for (int k0 = 0; k0 < Kpad; k0 += 32) {
        v8h alo = *(const v8h*)(Ap + k0);
        v8h ahi = *(const v8h*)(Ap + k0 + 8);
        v16h bv = *(const v16h*)(Bp + k0);
        acc = wmma_f16(cat8(alo, ahi), bv, acc);
    }
#pragma unroll
    for (int r = 0; r < 8; r++) {
        int mr = m0 + r + ((lane >> 4) << 3);
        if (ncol < Nn) {
            float v = acc[r] + (bias ? bias[ncol] : 0.f);
            if (act == 1) v = fmaxf(v, 0.f);
            else if (act == 2) v = sigf(v);
            if (C)  C [(size_t)mr * Nn + ncol] = v;
            if (Ch) Ch[(size_t)mr * Nn + ncol] = (_Float16)v;
        }
    }
}

// ---------------------------------------------------------------------------
// Persistent LSTM layer: 1 workgroup, 16 waves, Whh fragments in VGPRs,
// h state f16 in LDS, WMMA recurrence [32,128]@[128,512] per timestep.
// ---------------------------------------------------------------------------
__global__ __launch_bounds__(512) void lstm_seq_kernel(
    const _Float16* __restrict__ preh, const float* __restrict__ Whh,
    _Float16* __restrict__ houth)
{
    __shared__ _Float16 hbuf[32][Hd + 8];
    __shared__ _Float16 gstage[32][4 * Hd + 8];

    int tid  = threadIdx.x;
    int lane = tid & 31;
    int w    = tid >> 5;
    int m0   = (w & 1) * 16;
    int ng   = w >> 1;

    v16h Bf[4][4];
#pragma unroll
    for (int i = 0; i < 4; i++) {
        int n = (ng * 4 + i) * 16 + (lane & 15);
#pragma unroll
        for (int kk = 0; kk < 4; kk++) {
            v16h bv;
#pragma unroll
            for (int j = 0; j < 16; j++)
                bv[j] = (_Float16)Whh[(size_t)n * Hd + kk * 32 + ((lane >> 4) << 3) + (j & 7) + ((j >> 3) << 4)];
            Bf[i][kk] = bv;
        }
    }
    for (int idx = tid; idx < 32 * (Hd + 8); idx += 512)
        ((_Float16*)hbuf)[idx] = (_Float16)0.f;

    float cst[8];
#pragma unroll
    for (int p = 0; p < 8; p++) cst[p] = 0.f;
    int gb = tid >> 4;
    int gh = (tid & 15) * 8;
    __syncthreads();

    for (int t = 0; t < Ss; t++) {
        v16h Af[4];
        int mrow = m0 + (lane & 15);
#pragma unroll
        for (int kk = 0; kk < 4; kk++) {
            v8h lo = *(const v8h*)&hbuf[mrow][kk * 32 + ((lane >> 4) << 3)];
            v8h hi = *(const v8h*)&hbuf[mrow][kk * 32 + 16 + ((lane >> 4) << 3)];
            Af[kk] = cat8(lo, hi);
        }
#pragma unroll
        for (int i = 0; i < 4; i++) {
            int n0 = (ng * 4 + i) * 16;
            v8f c;
#pragma unroll
            for (int r = 0; r < 8; r++) {
                int bb = m0 + r + ((lane >> 4) << 3);
                c[r] = (float)preh[((size_t)bb * Ss + t) * (4 * Hd) + n0 + (lane & 15)];
            }
#pragma unroll
            for (int kk = 0; kk < 4; kk++) c = wmma_f16(Af[kk], Bf[i][kk], c);
#pragma unroll
            for (int r = 0; r < 8; r++) {
                int bb = m0 + r + ((lane >> 4) << 3);
                gstage[bb][n0 + (lane & 15)] = (_Float16)c[r];
            }
        }
        __syncthreads();
        {
            v8h gi = *(const v8h*)&gstage[gb][gh];
            v8h gf = *(const v8h*)&gstage[gb][Hd + gh];
            v8h gg = *(const v8h*)&gstage[gb][2 * Hd + gh];
            v8h go = *(const v8h*)&gstage[gb][3 * Hd + gh];
            v8h hv;
#pragma unroll
            for (int p = 0; p < 8; p++) {
                float cc = sigf((float)gf[p]) * cst[p] + sigf((float)gi[p]) * tanhf((float)gg[p]);
                cst[p] = cc;
                hv[p] = (_Float16)(sigf((float)go[p]) * tanhf(cc));
            }
            *(v8h*)&hbuf[gb][gh] = hv;
            *(v8h*)(houth + ((size_t)gb * Ss + t) * Hd + gh) = hv;
        }
        __syncthreads();
    }
}

// ---------------------------------------------------------------------------
// Flash attention over memory axis. Wave owns 16 query columns; Q fragments
// hoisted; K/V block TDM-staged into double-buffered LDS so the next block's
// DMA overlaps this block's WMMAs (TENSORcnt-tracked).
// ---------------------------------------------------------------------------
__global__ __launch_bounds__(256) void attn_kernel(
    const _Float16* __restrict__ memh, const _Float16* __restrict__ hidh,
    _Float16* __restrict__ outh)
{
    __shared__ _Float16 smem[2][32][Hd + 16];
    __shared__ float lbuf[8][16];
    int lane = threadIdx.x & 31;
    int w    = threadIdx.x >> 5;
    int b    = blockIdx.x >> 3;
    int tb   = blockIdx.x & 7;
    int t0   = tb * 128 + w * 16;
    const _Float16* Mg = memh + (size_t)b * Ss * Hd;
    const _Float16* Qg = hidh + (size_t)b * Ss * Hd;
    int tcol = t0 + (lane & 15);

    v16h Qf[4];
#pragma unroll
    for (int kk = 0; kk < 4; kk++)
        Qf[kk] = cat8(*(const v8h*)(Qg + (size_t)tcol * Hd + kk * 32 + ((lane >> 4) << 3)), *(const v8h*)(Qg + (size_t)tcol * Hd + kk * 32 + 16 + ((lane >> 4) << 3)));

    v8f o[8];
#pragma unroll
    for (int i = 0; i < 8; i++) o[i] = vzero8();
    float mrun = -3.0e38f, lrun = 0.f;

#if USE_TDM
    if (w == 0) tdm_load_rows(Mg, (unsigned)(size_t)&smem[0][0][0]);
#endif
    int buf = 0;
    for (int sb = 0; sb < Ss; sb += 32, buf ^= 1) {
#if USE_TDM
        if (w == 0) __builtin_amdgcn_s_wait_tensorcnt(0);
        __syncthreads();                       // data visible; prior reads done
        if (w == 0 && sb + 32 < Ss)
            tdm_load_rows(Mg + (size_t)(sb + 32) * Hd,
                          (unsigned)(size_t)&smem[buf ^ 1][0][0]);
#else
        __syncthreads();
        {
            int ldr = (threadIdx.x * 16) >> 7;
            int ldc = (threadIdx.x * 16) & 127;
            v8h x0 = *(const v8h*)(Mg + (size_t)(sb + ldr) * Hd + ldc);
            v8h x1 = *(const v8h*)(Mg + (size_t)(sb + ldr) * Hd + ldc + 8);
            *(v8h*)&smem[buf][ldr][ldc]     = x0;
            *(v8h*)&smem[buf][ldr][ldc + 8] = x1;
        }
        __syncthreads();
#endif
        const _Float16 (*sm)[Hd + 16] = smem[buf];

        v8f c0 = vzero8(), c1 = vzero8();
#pragma unroll
        for (int kk = 0; kk < 4; kk++) {
            int ko = kk * 32 + ((lane >> 4) << 3);
            v8h a0l = *(const v8h*)&sm[lane & 15][ko];
            v8h a0h = *(const v8h*)&sm[lane & 15][ko + 16];
            v8h a1l = *(const v8h*)&sm[16 + (lane & 15)][ko];
            v8h a1h = *(const v8h*)&sm[16 + (lane & 15)][ko + 16];
            c0 = wmma_f16(cat8(a0l, a0h), Qf[kk], c0);
            c1 = wmma_f16(cat8(a1l, a1h), Qf[kk], c1);
        }
        float mb = c0[0];
#pragma unroll
        for (int r = 0; r < 8; r++) { mb = fmaxf(mb, c0[r]); mb = fmaxf(mb, c1[r]); }
        mb = fmaxf(mb, __shfl_xor(mb, 16, 32));
        float mnew  = fmaxf(mrun, mb);
        float alpha = __expf(mrun - mnew);
        float ls = 0.f;
        v16h pa;                               // score C-layout == P^T A-layout
#pragma unroll
        for (int r = 0; r < 8; r++) {
            float p0 = __expf(c0[r] - mnew);
            float p1 = __expf(c1[r] - mnew);
            ls += p0 + p1;
            pa[r]     = (_Float16)p0;
            pa[8 + r] = (_Float16)p1;
        }
        ls += __shfl_xor(ls, 16, 32);
        lrun = lrun * alpha + ls;
        mrun = mnew;
#pragma unroll
        for (int d = 0; d < 8; d++)
#pragma unroll
            for (int r = 0; r < 8; r++) o[d][r] *= alpha;
#pragma unroll
        for (int d = 0; d < 8; d++) {
            v16h bv;                           // transposed V gather from LDS
#pragma unroll
            for (int j = 0; j < 16; j++)
                bv[j] = sm[((lane >> 4) << 3) + (j & 7) + ((j >> 3) << 4)][d * 16 + (lane & 15)];
            o[d] = wmma_f16(pa, bv, o[d]);
        }
    }
    if (lane < 16) lbuf[w][lane] = lrun;
    __syncthreads();
#pragma unroll
    for (int d = 0; d < 8; d++)
#pragma unroll
        for (int r = 0; r < 8; r++) {
            int rloc = r + ((lane >> 4) << 3);
            float v = o[d][r] / lbuf[w][rloc];
            outh[((size_t)b * Ss + t0 + rloc) * Hd + d * 16 + (lane & 15)] = (_Float16)v;
        }
}

// ---------------------------------------------------------------------------
// Prep / small kernels
// ---------------------------------------------------------------------------
__global__ void prep_weight_kernel(const float* __restrict__ W, int K, int N, int w_kn,
                                   _Float16* __restrict__ Wt, int Kpad, int Npad)
{
    int i = blockIdx.x * blockDim.x + threadIdx.x;
    if (i >= Npad * Kpad) return;
    int n = i / Kpad, k = i % Kpad;
    float v = 0.f;
    if (n < N && k < K) v = w_kn ? W[(size_t)k * N + n] : W[(size_t)n * K + k];
    Wt[i] = (_Float16)v;
}

__global__ void prep_input_kernel(const float* __restrict__ x, _Float16* __restrict__ xh)
{
    int i = blockIdx.x * blockDim.x + threadIdx.x;
    if (i >= Bb * Ss * 64) return;
    int r = i >> 6, k = i & 63;
    xh[i] = (_Float16)(k < INd ? x[(size_t)r * INd + k] : 0.f);
}

__global__ void vec_add_kernel(const float* a, const float* b, float* o, int n) {
    int i = blockIdx.x * blockDim.x + threadIdx.x;
    if (i < n) o[i] = a[i] + b[i];
}

__global__ void shift_target_kernel(const float* __restrict__ label, float* __restrict__ tgt) {
    int i = blockIdx.x * blockDim.x + threadIdx.x;
    if (i >= Bb * Ss * OUTd) return;
    int s = (i / OUTd) % Ss;
    tgt[i] = (s == 0) ? 0.f : label[i - OUTd];
}

__global__ void tgt_proj_kernel(const float* __restrict__ hid,
                                const float* __restrict__ Wte, const float* __restrict__ bte,
                                const float* __restrict__ Wpw, const float* __restrict__ bpw,
                                float* __restrict__ ht, float* __restrict__ memT)
{
    int i = blockIdx.x * blockDim.x + threadIdx.x;
    if (i >= Bb * Ss) return;
    const float* x = hid + (size_t)i * 3;
    float h[3];
#pragma unroll
    for (int e = 0; e < 3; e++)
        h[e] = bte[e] + x[0] * Wte[e] + x[1] * Wte[3 + e] + x[2] * Wte[6 + e];
#pragma unroll
    for (int e = 0; e < 3; e++) {
        float m = bpw[e] + h[0] * Wpw[e] + h[1] * Wpw[3 + e] + h[2] * Wpw[6 + e];
        memT[(size_t)i * 3 + e] = fmaxf(m, 0.f);
        ht[(size_t)i * 3 + e]   = h[e];
    }
}

__global__ void tgt_attn_kernel(const float* __restrict__ memT, const float* __restrict__ ht,
                                float* __restrict__ out)
{
    int i = blockIdx.x * blockDim.x + threadIdx.x;
    if (i >= Bb * Ss) return;
    int b = i / Ss, t = i % Ss;
    const float* q = ht + ((size_t)b * Ss + t) * 3;
    float q0 = q[0], q1 = q[1], q2 = q[2];
    float m = -3.0e38f, l = 0.f, a0 = 0.f, a1 = 0.f, a2 = 0.f;
    for (int s = 0; s <= t; s++) {
        const float* mp = memT + ((size_t)b * Ss + s) * 3;
        __builtin_prefetch(mp + 48, 0, 0);     // global_prefetch_b8, ~16 rows ahead
        float sc = mp[0] * q0 + mp[1] * q1 + mp[2] * q2;
        float mn = fmaxf(m, sc);
        float al = __expf(m - mn);
        float p  = __expf(sc - mn);
        l  = l * al + p;
        a0 = a0 * al + p * mp[0];
        a1 = a1 * al + p * mp[1];
        a2 = a2 * al + p * mp[2];
        m = mn;
    }
    float inv = 1.f / l;
    out[(size_t)i * 3 + 0] = a0 * inv;
    out[(size_t)i * 3 + 1] = a1 * inv;
    out[(size_t)i * 3 + 2] = a2 * inv;
}

__global__ void gate_kernel(const _Float16* __restrict__ mo, const float* __restrict__ mt,
                            const float* __restrict__ Wafm, const float* __restrict__ bafm,
                            float* __restrict__ zf, _Float16* __restrict__ zh)
{
    int i = blockIdx.x * blockDim.x + threadIdx.x;
    if (i >= Bb * Ss * Hd) return;
    int bs = i / Hd, h = i % Hd;
    const float* e = mt + (size_t)bs * 3;
    float map = bafm[h] + e[0] * Wafm[h] + e[1] * Wafm[Hd + h] + e[2] * Wafm[2 * Hd + h];
    float v = (float)mo[i] * sigf(map);
    size_t zi = (size_t)bs * (2 * Hd) + Hd + h;
    zf[zi] = v;
    zh[zi] = (_Float16)v;
}

__device__ __forceinline__ float hash_normal(unsigned long long idx) {
    unsigned long long z = idx * 0x9E3779B97F4A7C15ull + 0xBF58476D1CE4E5B9ull;
    z ^= z >> 30; z *= 0xBF58476D1CE4E5B9ull;
    z ^= z >> 27; z *= 0x94D049BB133111EBull;
    z ^= z >> 31;
    unsigned int u1 = (unsigned int)z, u2 = (unsigned int)(z >> 32);
    float f1 = (u1 + 0.5f) * (1.0f / 4294967296.0f);
    float f2 = u2 * (1.0f / 4294967296.0f);
    return sqrtf(-2.f * __logf(f1)) * __cosf(6.28318530718f * f2);
}

__global__ void z_kernel(const float* __restrict__ meanp, const float* __restrict__ stdp,
                         float* __restrict__ zf, _Float16* __restrict__ zh)
{
    int i = blockIdx.x * blockDim.x + threadIdx.x;
    if (i >= Bb * Ss * Hd) return;
    int bs = i / Hd, h = i % Hd;
    float v = hash_normal((unsigned long long)i) * stdp[i] + meanp[i];
    size_t zi = (size_t)bs * (2 * Hd) + h;
    zf[zi] = v;
    zh[zi] = (_Float16)v;
}

// ---------------------------------------------------------------------------
extern "C" void kernel_launch(void* const* d_in, const int* in_sizes, int n_in,
                              void* d_out, int out_size, void* d_ws, size_t ws_size,
                              hipStream_t stream)
{
    const float* input = (const float*)d_in[0];
    const float* label = (const float*)d_in[1];
    const float* Wih0  = (const float*)d_in[2];
    const float* Whh0  = (const float*)d_in[3];
    const float* bih0  = (const float*)d_in[4];
    const float* bhh0  = (const float*)d_in[5];
    const float* Wih1  = (const float*)d_in[6];
    const float* Whh1  = (const float*)d_in[7];
    const float* bih1  = (const float*)d_in[8];
    const float* bhh1  = (const float*)d_in[9];
    const float* Wmw = (const float*)d_in[10]; const float* bmw = (const float*)d_in[11];
    const float* Wte = (const float*)d_in[12]; const float* bte = (const float*)d_in[13];
    const float* Wpw = (const float*)d_in[14]; const float* bpw = (const float*)d_in[15];
    const float* Wafm = (const float*)d_in[16]; const float* bafm = (const float*)d_in[17];
    const float* Wmean = (const float*)d_in[18]; const float* bmean = (const float*)d_in[19];
    const float* Wstd = (const float*)d_in[20]; const float* bstd = (const float*)d_in[21];
    const float* Wout = (const float*)d_in[22]; const float* bout = (const float*)d_in[23];
    const float* Wmo  = (const float*)d_in[24]; const float* bmo  = (const float*)d_in[25];

    float* ws = (float*)d_ws;
    const size_t BS   = (size_t)Bb * Ss;
    const size_t nBSH = BS * Hd;
    const size_t nBSE = BS * 3;

    size_t off = 0;
    float* bias0  = ws + off; off += 512;
    float* bias1  = ws + off; off += 512;
    float* target = ws + off; off += nBSE;
    float* htb    = ws + off; off += nBSE;
    float* memT   = ws + off; off += nBSE;
    float* tgtA   = ws + off; off += nBSE;
    float* tgtB   = ws + off; off += nBSE;
    _Float16* Wih0t  = (_Float16*)(ws + off); off += 16384;
    _Float16* Wih1t  = (_Float16*)(ws + off); off += 32768;
    _Float16* Wmwt   = (_Float16*)(ws + off); off += 8192;
    _Float16* Wmeant = (_Float16*)(ws + off); off += 8192;
    _Float16* Wstdt  = (_Float16*)(ws + off); off += 8192;
    _Float16* Woutt  = (_Float16*)(ws + off); off += 2048;
    _Float16* Wmot   = (_Float16*)(ws + off); off += 3072;
    _Float16* inh    = (_Float16*)(ws + off); off += 1048576;
    _Float16* h1h    = (_Float16*)(ws + off); off += nBSH / 2;
    _Float16* h2h    = (_Float16*)(ws + off); off += nBSH / 2;
    _Float16* memh   = (_Float16*)(ws + off); off += nBSH / 2;
    _Float16* bufAh  = (_Float16*)(ws + off); off += nBSH / 2;
    _Float16* bufBh  = (_Float16*)(ws + off); off += nBSH / 2;
    _Float16* preh   = (_Float16*)(ws + off); off += BS * 4 * Hd / 2;
    _Float16* zh     = (_Float16*)(ws + off); off += BS * 2 * Hd / 2;

    float* o_x    = (float*)d_out;
    float* o_z    = o_x + nBSE;
    float* o_mean = o_z + BS * 2 * Hd;
    float* o_std  = o_mean + nBSH;
    float* o_mp   = o_std + nBSH;

    auto cdiv = [](long long n, long long d) { return (int)((n + d - 1) / d); };
    auto gemm_blocks = [&](int M, int N) {
        return cdiv((long long)(M >> 4) * cdiv(N, 16), 4);
    };

    vec_add_kernel<<<2, 256, 0, stream>>>(bih0, bhh0, bias0, 512);
    vec_add_kernel<<<2, 256, 0, stream>>>(bih1, bhh1, bias1, 512);
    shift_target_kernel<<<cdiv(nBSE, 256), 256, 0, stream>>>(label, target);
    prep_input_kernel<<<cdiv(BS * 64, 256), 256, 0, stream>>>(input, inh);
    prep_weight_kernel<<<cdiv(512 * 64, 256), 256, 0, stream>>>(Wih0, INd, 512, 0, Wih0t, 64, 512);
    prep_weight_kernel<<<cdiv(512 * 128, 256), 256, 0, stream>>>(Wih1, Hd, 512, 0, Wih1t, 128, 512);
    prep_weight_kernel<<<cdiv(128 * 128, 256), 256, 0, stream>>>(Wmw, Hd, Hd, 1, Wmwt, 128, 128);
    prep_weight_kernel<<<cdiv(128 * 128, 256), 256, 0, stream>>>(Wmean, Hd, Hd, 1, Wmeant, 128, 128);
    prep_weight_kernel<<<cdiv(128 * 128, 256), 256, 0, stream>>>(Wstd, Hd, Hd, 1, Wstdt, 128, 128);
    prep_weight_kernel<<<cdiv(16 * 256, 256), 256, 0, stream>>>(Wout, 2 * Hd, OUTd, 1, Woutt, 256, 16);
    prep_weight_kernel<<<cdiv(48 * 128, 256), 256, 0, stream>>>(Wmo, Hd, INd, 1, Wmot, 128, 48);

    // ---- LSTM layers ----
    wmma_gemm_kernel<<<gemm_blocks((int)BS, 512), 128, 0, stream>>>(
        inh, 64, Wih0t, 64, bias0, nullptr, preh, (int)BS, 512, 64, 0);
    lstm_seq_kernel<<<1, 512, 0, stream>>>(preh, Whh0, h1h);
    wmma_gemm_kernel<<<gemm_blocks((int)BS, 512), 128, 0, stream>>>(
        h1h, 128, Wih1t, 128, bias1, nullptr, preh, (int)BS, 512, 128, 0);
    lstm_seq_kernel<<<1, 512, 0, stream>>>(preh, Whh1, h2h);

    // ---- memnet hops (WMMA flash attention, TDM-staged K/V) ----
    wmma_gemm_kernel<<<gemm_blocks((int)BS, Hd), 128, 0, stream>>>(
        h2h, 128, Wmwt, 128, bmw, nullptr, memh, (int)BS, Hd, 128, 1);
    attn_kernel<<<Bb * 8, 256, 0, stream>>>(memh, h2h, bufAh);
    wmma_gemm_kernel<<<gemm_blocks((int)BS, Hd), 128, 0, stream>>>(
        bufAh, 128, Wmwt, 128, bmw, nullptr, memh, (int)BS, Hd, 128, 1);
    attn_kernel<<<Bb * 8, 256, 0, stream>>>(memh, bufAh, bufBh);

    // ---- target memnet (E=3), 2 hops, causal ----
    tgt_proj_kernel<<<cdiv(BS, 256), 256, 0, stream>>>(target, Wte, bte, Wpw, bpw, htb, memT);
    tgt_attn_kernel<<<cdiv(BS, 256), 256, 0, stream>>>(memT, htb, tgtA);
    tgt_proj_kernel<<<cdiv(BS, 256), 256, 0, stream>>>(tgtA, Wte, bte, Wpw, bpw, htb, memT);
    tgt_attn_kernel<<<cdiv(BS, 256), 256, 0, stream>>>(memT, htb, tgtB);

    // ---- gating, heads, z, outputs ----
    gate_kernel<<<cdiv(nBSH, 256), 256, 0, stream>>>(bufBh, tgtB, Wafm, bafm, o_z, zh);
    wmma_gemm_kernel<<<gemm_blocks((int)BS, Hd), 128, 0, stream>>>(
        h2h, 128, Wmeant, 128, bmean, o_mean, nullptr, (int)BS, Hd, 128, 2);
    wmma_gemm_kernel<<<gemm_blocks((int)BS, Hd), 128, 0, stream>>>(
        h2h, 128, Wstdt, 128, bstd, o_std, nullptr, (int)BS, Hd, 128, 1);
    z_kernel<<<cdiv(nBSH, 256), 256, 0, stream>>>(o_mean, o_std, o_z, zh);
    wmma_gemm_kernel<<<gemm_blocks((int)BS, OUTd), 128, 0, stream>>>(
        zh, 256, Woutt, 256, bout, o_x, nullptr, (int)BS, OUTd, 256, 0);
    wmma_gemm_kernel<<<gemm_blocks((int)BS, INd), 128, 0, stream>>>(
        zh + Hd, 256, Wmot, 128, bmo, o_mp, nullptr, (int)BS, INd, 128, 0);
}
